// GestureNet_58360015618609
// MI455X (gfx1250) — matrix operand order repaired and checked by reading, not compiled, not run on hardware
//
#include <hip/hip_runtime.h>

typedef __attribute__((ext_vector_type(16))) _Float16 v16h;
typedef __attribute__((ext_vector_type(8)))  _Float16 v8h;
typedef __attribute__((ext_vector_type(8)))  float    v8f;

// CDNA5 async global->LDS path (gfx1250), guarded so older toolchains fall
// back to load + ds_store staging.
#if defined(__has_builtin)
#  if __has_builtin(__builtin_amdgcn_global_load_async_to_lds_b128) && \
      __has_builtin(__builtin_amdgcn_s_wait_asynccnt)
#    define USE_ASYNC_LDS 1
#  endif
#endif
#ifndef USE_ASYNC_LDS
#  define USE_ASYNC_LDS 0
#endif

#if USE_ASYNC_LDS
// builtin signature: (v4i addrspace(1)* src, v4i addrspace(3)* dst, Ii, Ii)
typedef int vi4 __attribute__((vector_size(4 * sizeof(int))));
typedef __attribute__((address_space(1))) vi4* gvi4p;
typedef __attribute__((address_space(3))) vi4* lvi4p;
#endif

// ---- SLAYER constants (tau=5, theta=1.25, Ts=1) ----
static constexpr float D_     = 0.81873075307798182f;   // exp(-1/5)
static constexpr float PSPS_  = 0.54365636569180902f;   // e/5
static constexpr float THETA_ = 1.25f;
static constexpr float REFS_  = 2.0f * 1.25f * 0.54365636569180902f; // 2*theta*e/5
static constexpr int   TT     = 300;
static constexpr int   BB     = 4;

// one fused psp+spike timestep; IIR state carried in VGPRs
__device__ __forceinline__ float psp_spike_step(float x, float& s1, float& s2,
                                                float& r1, float& r2) {
    s2 = D_ * (s2 + s1);          // psp alpha state
    s1 = D_ * s1 + x;
    r2 = D_ * (r2 + r1);          // refractory alpha state
    r1 = D_ * r1;
    float p  = PSPS_ * s2;
    float sv = (p - REFS_ * r2 - THETA_ >= 0.0f) ? 1.0f : 0.0f;
    r1 += sv;
    return sv;
}

// ---------------------------------------------------------------------------
// Weight pre-convert: f32 [O,Kreal] -> f16 [Opad,Kpad], zero padded.
// ---------------------------------------------------------------------------
__global__ void cvt_w_kernel(const float* __restrict__ w, _Float16* __restrict__ wf,
                             int O, int Kreal, int Opad, int Kpad) {
    int idx = blockIdx.x * blockDim.x + threadIdx.x;
    if (idx >= Opad * Kpad) return;
    int o = idx / Kpad, k = idx - o * Kpad;
    wf[idx] = (o < O && k < Kreal) ? (_Float16)w[(size_t)o * Kreal + k]
                                   : (_Float16)0.0f;
}

// ---------------------------------------------------------------------------
// K0: 4x4 sum-pool (*1.1*theta) + psp + spike.  s_in [B,2,128,128,T] (T last)
//     -> sp [B,T,2*32*32] f16, frame-major
// ---------------------------------------------------------------------------
__global__ void pool4_scan_kernel(const float* __restrict__ s_in,
                                  _Float16* __restrict__ sp) {
    int idx = blockIdx.x * blockDim.x + threadIdx.x;
    if (idx >= BB * 2 * 32 * 32) return;
    int b  = idx >> 11;
    int r  = idx & 2047;
    int c  = r >> 10;
    int rr = r & 1023;
    int py = rr >> 5, px = rr & 31;

    const float* base = s_in + (size_t)(b * 2 + c) * 128 * 128 * TT;
    int outoff = c * 1024 + py * 32 + px;
    float s1 = 0.f, s2 = 0.f, r1 = 0.f, r2 = 0.f;
    for (int t = 0; t < TT; ++t) {
        float sum = 0.f;
#pragma unroll
        for (int dy = 0; dy < 4; ++dy)
#pragma unroll
            for (int dx = 0; dx < 4; ++dx)
                sum += base[(size_t)((4 * py + dy) * 128 + 4 * px + dx) * TT + t];
        float sv = psp_spike_step(1.375f * sum, s1, s2, r1, r2);
        sp[(size_t)(b * TT + t) * 2048 + outoff] = (_Float16)sv;
    }
}

// ---------------------------------------------------------------------------
// Conv as per-frame GEMM on v_wmma_f32_16x16x32_f16, frame staged in LDS
// (async global->LDS when available).
//   A (M=O x K=Cin*k*k): f16 weights, K zero-padded to Kpad (no guards).
//   B (K x N=H*H pixels): im2col gathered from LDS (ds_load_u16).
//   Fragment k order contiguous: lane half h holds k = 16h..16h+15 (wave32).
//   u out: [frame, O, H*H] f32 (frame-major).
// ---------------------------------------------------------------------------
template <int Cin, int H, int O, int KSZ, int PAD, int WPB>
__global__ void conv_wmma_t(const _Float16* __restrict__ sp,
                            const _Float16* __restrict__ wf,
                            float* __restrict__ u) {
    constexpr int HW    = H * H;
    constexpr int Kreal = Cin * KSZ * KSZ;
    constexpr int Kpad  = ((Kreal + 31) / 32) * 32;
    constexpr int KCH   = Kpad / 32;
    constexpr int MT    = O / 16;
    constexpr int NT    = HW / 16;
    constexpr int TILES = MT * NT;
    constexpr int BPF   = (TILES + WPB - 1) / WPB;  // blocks per frame
    constexpr int FRAME = Cin * HW;                 // halves per frame

    const int frame = blockIdx.x / BPF;
    const int tb    = blockIdx.x % BPF;

    __shared__ _Float16 lsp[FRAME];
    {   // cooperative 128-bit staging of this frame's spikes into LDS
        const uint4* g = (const uint4*)(sp + (size_t)frame * FRAME);
        uint4*       l = (uint4*)lsp;
#if USE_ASYNC_LDS
        for (int i = threadIdx.x; i < FRAME / 8; i += WPB * 32)
            __builtin_amdgcn_global_load_async_to_lds_b128(
                (gvi4p)(g + i), (lvi4p)(l + i), 0, 0);
        __builtin_amdgcn_s_wait_asynccnt(0);
#else
        for (int i = threadIdx.x; i < FRAME / 8; i += WPB * 32) l[i] = g[i];
#endif
    }
    __syncthreads();

    const int wave = threadIdx.x >> 5;
    const int tile = tb * WPB + wave;
    if (tile >= TILES) return;                 // wave-uniform
    const int lane  = threadIdx.x & 31;
    const int mtile = tile / NT;
    const int ntile = tile - mtile * NT;

    // A-fragment (16-bit A 16x32): lanes 0-15 K=0..15, lanes 16-31 K=16..31
    const int mrow   = lane & 15;
    const int kbaseA = (lane < 16) ? 0 : 16;
    const _Float16* wrow = wf + (size_t)(mtile * 16 + mrow) * Kpad;
    // B-fragment (16-bit B 32x16): lanes 0-15 K=0..15, lanes 16-31 K=16..31
    const int ncol   = lane & 15;
    const int pix    = ntile * 16 + ncol;
    const int oy     = pix / H, ox = pix - (pix / H) * H;
    const int kbaseB = (lane < 16) ? 0 : 16;

    v8f acc = {};
#pragma unroll
    for (int kc = 0; kc < KCH; ++kc) {
        const int k0 = kc << 5;
        v8h alo = *(const v8h*)(wrow + k0 + kbaseA);
        v8h ahi = *(const v8h*)(wrow + k0 + kbaseA + 8);
        v16h a  = __builtin_shufflevector(alo, ahi, 0, 1, 2, 3, 4, 5, 6, 7,
                                          8, 9, 10, 11, 12, 13, 14, 15);
        v16h b;
#pragma unroll
        for (int e = 0; e < 16; ++e) {
            const int kb = k0 + kbaseB + e;
            _Float16 bv = (_Float16)0.0f;
            if ((Kreal == Kpad) || (kb < Kreal)) {   // folds when K is full
                int c  = kb / (KSZ * KSZ);
                int r  = kb - c * (KSZ * KSZ);
                int kh = r / KSZ;
                int kw = r - kh * KSZ;
                int iy = oy + kh - PAD;
                int ix = ox + kw - PAD;
                if ((unsigned)iy < (unsigned)H && (unsigned)ix < (unsigned)H)
                    bv = lsp[c * HW + iy * H + ix];
            }
            b[e] = bv;
        }
        acc = __builtin_amdgcn_wmma_f32_16x16x32_f16(false, a, false, b,
                                                     (short)0, acc, false, false);
    }

    // C layout: lane<16 -> rows 0-7, lane>=16 -> rows 8-15; col = lane&15
    float* uf = u + (size_t)frame * O * HW;
    const int rbase = (lane < 16) ? 0 : 8;
#pragma unroll
    for (int r = 0; r < 8; ++r)
        uf[(size_t)(mtile * 16 + rbase + r) * HW + pix] = acc[r];
}

// ---------------------------------------------------------------------------
// Dense GEMM over N = B*T columns. x [B,T,F] f16 (column (b,t) K-contiguous:
// B-fragment = one 32-byte contiguous v16h load).  wf [Opad,F] f16.
// u out: [B,T,Opad] f32.
// ---------------------------------------------------------------------------
template <int F, int Opad, int WPB>
__global__ void dense_wmma_t(const _Float16* __restrict__ x,
                             const _Float16* __restrict__ wf,
                             float* __restrict__ u) {
    constexpr int MT  = Opad / 16;
    constexpr int NTL = (BB * TT) / 16;  // 75, exact
    const int gw = blockIdx.x * WPB + (threadIdx.x >> 5);
    if (gw >= MT * NTL) return;          // wave-uniform
    const int lane  = threadIdx.x & 31;
    const int ntile = gw % NTL;
    const int mtile = gw / NTL;

    const int mrow   = lane & 15;
    const int kbaseA = (lane < 16) ? 0 : 16;
    const _Float16* wrow = wf + (size_t)(mtile * 16 + mrow) * F;
    const int col    = ntile * 16 + (lane & 15);
    const int kbaseB = (lane < 16) ? 0 : 16;
    const _Float16* xc = x + (size_t)col * F;

    v8f acc = {};
    for (int kc = 0; kc < F / 32; ++kc) {
        const int k0 = kc << 5;
        v8h alo = *(const v8h*)(wrow + k0 + kbaseA);
        v8h ahi = *(const v8h*)(wrow + k0 + kbaseA + 8);
        v16h a  = __builtin_shufflevector(alo, ahi, 0, 1, 2, 3, 4, 5, 6, 7,
                                          8, 9, 10, 11, 12, 13, 14, 15);
        v16h b  = *(const v16h*)(xc + k0 + kbaseB);
        acc = __builtin_amdgcn_wmma_f32_16x16x32_f16(false, a, false, b,
                                                     (short)0, acc, false, false);
    }

    const int rbase = (lane < 16) ? 0 : 8;
#pragma unroll
    for (int r = 0; r < 8; ++r)
        u[(size_t)col * Opad + mtile * 16 + rbase + r] = acc[r];
}

// ---------------------------------------------------------------------------
// psp + spike scan over T.  u [B,T,N] f32 -> sp [B,T,N] f16.  Thread per
// neuron; per-t accesses contiguous across adjacent lanes.
// ---------------------------------------------------------------------------
__global__ void scan_kernel(const float* __restrict__ u,
                            _Float16* __restrict__ sp, int N, int total) {
    int idx = blockIdx.x * blockDim.x + threadIdx.x;
    if (idx >= total) return;
    int b = idx / N;
    int i = idx - b * N;
    float s1 = 0.f, s2 = 0.f, r1 = 0.f, r2 = 0.f;
    for (int t = 0; t < TT; ++t) {
        size_t off = (size_t)(b * TT + t) * N + i;
        if (t + 8 < TT) __builtin_prefetch(&u[off + (size_t)8 * N], 0, 1);
        float sv = psp_spike_step(u[off], s1, s2, r1, r2);
        sp[off] = (_Float16)sv;
    }
}

// ---------------------------------------------------------------------------
// 2x2 sum-pool (*1.1*theta) + psp + spike.
// spin [B,T,C*H*H] -> spout [B,T,C*(H/2)*(H/2)]
// ---------------------------------------------------------------------------
__global__ void pool2_scan_kernel(const _Float16* __restrict__ spin,
                                  _Float16* __restrict__ spout,
                                  int C, int H, int total) {
    int idx = blockIdx.x * blockDim.x + threadIdx.x;
    if (idx >= total) return;
    int Ho = H >> 1;
    int No = C * Ho * Ho;
    int Ni = C * H * H;
    int b = idx / No;
    int i = idx - b * No;
    int c = i / (Ho * Ho);
    int r = i - c * Ho * Ho;
    int py = r / Ho, px = r - (r / Ho) * Ho;
    int in0 = c * H * H + (2 * py) * H + 2 * px;

    float s1 = 0.f, s2 = 0.f, r1 = 0.f, r2 = 0.f;
    for (int t = 0; t < TT; ++t) {
        const _Float16* f = spin + (size_t)(b * TT + t) * Ni;
        float sum = (float)f[in0] + (float)f[in0 + 1] +
                    (float)f[in0 + H] + (float)f[in0 + H + 1];
        float sv = psp_spike_step(1.375f * sum, s1, s2, r1, r2);
        spout[(size_t)(b * TT + t) * No + i] = (_Float16)sv;
    }
}

// ---------------------------------------------------------------------------
// Final scan: u [B,T,16] (11 valid) -> out [B,11,T] f32 (reference layout)
// ---------------------------------------------------------------------------
__global__ void final_scan_kernel(const float* __restrict__ u,
                                  float* __restrict__ out) {
    int idx = blockIdx.x * blockDim.x + threadIdx.x;
    if (idx >= BB * 11) return;
    int b = idx / 11, o = idx - (idx / 11) * 11;
    float s1 = 0.f, s2 = 0.f, r1 = 0.f, r2 = 0.f;
    for (int t = 0; t < TT; ++t) {
        float x  = u[(size_t)(b * TT + t) * 16 + o];
        float sv = psp_spike_step(x, s1, s2, r1, r2);
        out[(size_t)(b * 11 + o) * TT + t] = sv;
    }
}

// ---------------------------------------------------------------------------
extern "C" void kernel_launch(void* const* d_in, const int* in_sizes, int n_in,
                              void* d_out, int out_size, void* d_ws, size_t ws_size,
                              hipStream_t stream) {
    const float* s_in = (const float*)d_in[0];
    const float* w1   = (const float*)d_in[1];   // [32,2,5,5]
    const float* w2   = (const float*)d_in[2];   // [64,32,3,3]
    const float* w3   = (const float*)d_in[3];   // [128,64,3,3]
    const float* w4a  = (const float*)d_in[4];   // [256,2048]
    const float* w4b  = (const float*)d_in[5];   // [11,256]

    char* ws = (char*)d_ws;
    float*    u   = (float*)ws;                                   // <= 157.3 MB
    _Float16* spA = (_Float16*)(ws + (size_t)158 * 1024 * 1024);  // <= 19.7 MB
    _Float16* spB = (_Float16*)(ws + (size_t)178 * 1024 * 1024);  // <= 78.7 MB
    char*     wb  = ws + (size_t)258 * 1024 * 1024;               // f16 weights
    _Float16* wf1  = (_Float16*)(wb + 0);        // [32,64]
    _Float16* wf2  = (_Float16*)(wb + 8192);     // [64,288]
    _Float16* wf3  = (_Float16*)(wb + 45056);    // [128,576]
    _Float16* wf4a = (_Float16*)(wb + 192512);   // [256,2048]
    _Float16* wf4b = (_Float16*)(wb + 1241088);  // [16,256]

    const int frames = BB * TT;  // 1200
    const int THR = 256;         // 8 waves/block (scan/cvt kernels)
    const int CTHR = 512;        // 16 waves/block (conv kernels)
    auto blk1 = [](int n) { return (n + 255) / 256; };

    // --- weights f32 -> f16 (K padded to mult. of 32, O padded for dense2) ---
    cvt_w_kernel<<<blk1(32 * 64), THR, 0, stream>>>(w1, wf1, 32, 50, 32, 64);
    cvt_w_kernel<<<blk1(64 * 288), THR, 0, stream>>>(w2, wf2, 64, 288, 64, 288);
    cvt_w_kernel<<<blk1(128 * 576), THR, 0, stream>>>(w3, wf3, 128, 576, 128, 576);
    cvt_w_kernel<<<blk1(256 * 2048), THR, 0, stream>>>(w4a, wf4a, 256, 2048, 256, 2048);
    cvt_w_kernel<<<blk1(16 * 256), THR, 0, stream>>>(w4b, wf4b, 11, 256, 16, 256);

    // L0: pool4 + scan : s_in -> spA [B,T,2048]
    pool4_scan_kernel<<<blk1(BB * 2048), THR, 0, stream>>>(s_in, spA);

    // conv1 (O=32,Cin=2,k=5,p=2,H=32): 128 tiles/frame, 8 blocks/frame
    conv_wmma_t<2, 32, 32, 5, 2, 16><<<frames * 8, CTHR, 0, stream>>>(spA, wf1, u);
    scan_kernel<<<blk1(BB * 32768), THR, 0, stream>>>(u, spB, 32768, BB * 32768);
    pool2_scan_kernel<<<blk1(BB * 8192), THR, 0, stream>>>(spB, spA, 32, 32, BB * 8192);

    // conv2 (O=64,Cin=32,k=3,p=1,H=16): 64 tiles/frame, 4 blocks/frame
    conv_wmma_t<32, 16, 64, 3, 1, 16><<<frames * 4, CTHR, 0, stream>>>(spA, wf2, u);
    scan_kernel<<<blk1(BB * 16384), THR, 0, stream>>>(u, spB, 16384, BB * 16384);
    pool2_scan_kernel<<<blk1(BB * 4096), THR, 0, stream>>>(spB, spA, 64, 16, BB * 4096);

    // conv3 (O=128,Cin=64,k=3,p=1,H=8): 32 tiles/frame, 2 blocks/frame
    conv_wmma_t<64, 8, 128, 3, 1, 16><<<frames * 2, CTHR, 0, stream>>>(spA, wf3, u);
    scan_kernel<<<blk1(BB * 8192), THR, 0, stream>>>(u, spB, 8192, BB * 8192);
    pool2_scan_kernel<<<blk1(BB * 2048), THR, 0, stream>>>(spB, spA, 128, 8, BB * 2048);

    // dense1 (256 x 2048): 16*75 = 1200 waves -> 150 blocks
    dense_wmma_t<2048, 256, 8><<<150, THR, 0, stream>>>(spA, wf4a, u);
    scan_kernel<<<blk1(BB * 256), THR, 0, stream>>>(u, spB, 256, BB * 256);

    // dense2 (11 -> pad 16 x 256): 75 waves -> 10 blocks
    dense_wmma_t<256, 16, 8><<<10, THR, 0, stream>>>(spB, wf4b, u);

    // final scan -> d_out [B,11,T] f32
    final_scan_kernel<<<blk1(BB * 11), THR, 0, stream>>>(u, (float*)d_out);
}
